// GraphEmbeddingModel_53060025975546
// MI455X (gfx1250) — matrix operand order repaired and checked by reading, not compiled, not run on hardware
//
#include <hip/hip_runtime.h>
#include <math.h>

#define D   128
#define H   256
#define K1  384
#define BLOCK 256
#define USE_TDM 1
#define USE_ASYNC 1

typedef __attribute__((ext_vector_type(16))) __bf16 v16bf;
typedef __attribute__((ext_vector_type(8)))  float  v8f;
typedef __attribute__((ext_vector_type(4)))  unsigned v4u;
typedef __attribute__((ext_vector_type(8)))  unsigned v8u;

union Frag { v16bf v; unsigned u[8]; };

__device__ __forceinline__ unsigned short f32_to_bf16(float f) {
    union { float f; unsigned u; } c; c.f = f;
    unsigned r = 0x7FFFu + ((c.u >> 16) & 1u);
    return (unsigned short)((c.u + r) >> 16);
}
__device__ __forceinline__ unsigned pack2(float a, float b) {
    return (unsigned)f32_to_bf16(a) | ((unsigned)f32_to_bf16(b) << 16);
}
__device__ __forceinline__ float tanh_fast(float x) {
#if defined(__has_builtin) && __has_builtin(__builtin_amdgcn_tanhf)
    return __builtin_amdgcn_tanhf(x);          // native V_TANH_F32 on gfx1250
#else
    float e = __expf(-2.0f * fabsf(x));
    float t = (1.0f - e) / (1.0f + e);
    return copysignf(t, x);
#endif
}
__device__ __forceinline__ float gelu_tanh(float x) {
    return 0.5f * x * (1.0f + tanh_fast(0.7978845608028654f * (x + 0.044715f * x * x * x)));
}

// ---------------------------------------------------------------------------
// TDM: 1-D DMA of n8 8-byte elements from global into LDS (weight staging).
// ---------------------------------------------------------------------------
__device__ __forceinline__ void tdm_load_1d(unsigned lds_addr, const void* gsrc,
                                            unsigned n8) {
    unsigned long long ga = (unsigned long long)gsrc;
    v4u g0;
    g0[0] = 1u;                                            // count=1, user mode
    g0[1] = lds_addr;                                      // LDS byte address
    g0[2] = (unsigned)ga;                                  // global_addr[31:0]
    g0[3] = (unsigned)((ga >> 32) & 0x01FFFFFFu) | (2u << 30); // addr[56:32], type=2
    v8u g1;
    g1[0] = 3u << 16;                                      // data_size = 8 bytes
    g1[1] = (n8 & 0xFFFFu) << 16;                          // tensor_dim0[15:0]
    g1[2] = ((n8 >> 16) & 0xFFFFu) | (1u << 16);           // dim0[31:16], dim1=1
    g1[3] = (n8 & 0xFFFFu) << 16;                          // dim1 hi=0, tile_dim0
    g1[4] = 1u;                                            // tile_dim1=1
    g1[5] = n8;                                            // dim0_stride[31:0]
    g1[6] = 0u;
    g1[7] = 0u;
    asm volatile("tensor_load_to_lds %0, %1" :: "s"(g0), "s"(g1) : "memory");
}

__device__ __forceinline__ unsigned lds_addr_of(const void* p) {
    return (unsigned)(unsigned long long)p;   // low 32 bits = LDS byte offset
}

// Async global->LDS 16-byte copy (ASYNCcnt-tracked), saddr form.
__device__ __forceinline__ void async_gather_b128(unsigned lds_dst, unsigned voff,
                                                  unsigned long long sbase) {
#if USE_ASYNC
    asm volatile("global_load_async_to_lds_b128 %0, %1, %2"
                 :: "v"(lds_dst), "v"(voff), "s"(sbase) : "memory");
#endif
}

// ---------------------------------------------------------------------------
// One-time prep: pack f32 weight [K][N] into bf16 (k,k+1)-pair dwords [K/2][N].
// ---------------------------------------------------------------------------
__global__ void pack_weights_kernel(const float* __restrict__ w,
                                    unsigned* __restrict__ out, int k2n, int nn) {
    int i = blockIdx.x * blockDim.x + threadIdx.x;
    int stride = gridDim.x * blockDim.x;
    for (; i < k2n; i += stride) {
        int k2 = i / nn, c = i - k2 * nn;
        out[i] = pack2(w[(size_t)(2 * k2) * nn + c], w[(size_t)(2 * k2 + 1) * nn + c]);
    }
}

// ---------------------------------------------------------------------------
// Embedding: out = LayerNorm(gelu(in @ W + b)); writes f32 (optional) + bf16
// ---------------------------------------------------------------------------
__global__ __launch_bounds__(BLOCK)
void embed_ln_kernel(const float* __restrict__ in, const unsigned* __restrict__ wpk,
                     const float* __restrict__ bias, const float* __restrict__ ln_s,
                     const float* __restrict__ ln_o, float* __restrict__ out_f32,
                     unsigned short* __restrict__ out_bf, int rows)
{
    __shared__ unsigned       wp[(D / 2) * D];     // 32 KB packed bf16 pairs
    __shared__ unsigned short a_lds[16 * D];
    __shared__ float          o_lds[16 * D];
    __shared__ float          b_lds[D];

    const int tid = threadIdx.x;
#if USE_TDM
    if (tid < 32) {
        tdm_load_1d(lds_addr_of(wp), wpk, (D / 2) * D / 2);
        __builtin_amdgcn_s_wait_tensorcnt(0);
    }
#else
    for (int p = tid; p < (D / 2) * D; p += BLOCK) wp[p] = wpk[p];
#endif
    if (tid < D) b_lds[tid] = bias[tid];
    __syncthreads();

    const int wave = tid >> 5, lane = tid & 31, hi = lane >> 4, mrow = lane & 15;
    const int col = wave * 16 + (lane & 15);
    const int ntiles = (rows + 15) >> 4;

    for (int tile = blockIdx.x; tile < ntiles; tile += gridDim.x) {
        const int row0 = tile << 4;
        for (int p = tid; p < 16 * D; p += BLOCK) {
            int r = p >> 7, c = p & (D - 1);
            float v = (row0 + r < rows) ? in[(size_t)(row0 + r) * D + c] : 0.0f;
            a_lds[p] = f32_to_bf16(v);
        }
        __syncthreads();

        v8f acc;
        float bb = b_lds[col];
        #pragma unroll
        for (int r = 0; r < 8; ++r) acc[r] = bb;

        #pragma unroll
        for (int kb = 0; kb < D / 32; ++kb) {
            const int k0 = kb * 32;
            Frag a, b;
            #pragma unroll
            for (int q = 0; q < 8; ++q) {
                int ka = k0 + 2 * q + ((q >= 4) ? 8 : 0) + 8 * hi;
                a.u[q] = *reinterpret_cast<const unsigned*>(&a_lds[mrow * D + ka]);
                b.u[q] = wp[((k0 >> 1) + q + ((q >= 4) ? 4 : 0) + 4 * hi) * D + col];
            }
            acc = __builtin_amdgcn_wmma_f32_16x16x32_bf16(false, a.v, false, b.v,
                                                          (short)0, acc, false, false);
        }
        #pragma unroll
        for (int r = 0; r < 8; ++r) {
            int row = hi ? r + 8 : r;
            o_lds[row * D + col] = gelu_tanh(acc[r]);
        }
        __syncthreads();

        {   // LayerNorm: 16 threads per row, 8 elements each
            int r = tid >> 4, sub = tid & 15;
            float s = 0.0f, s2 = 0.0f;
            #pragma unroll
            for (int j = 0; j < 8; ++j) {
                float v = o_lds[r * D + sub * 8 + j];
                s += v; s2 += v * v;
            }
            #pragma unroll
            for (int m = 8; m >= 1; m >>= 1) {
                s  += __shfl_xor(s,  m, 16);
                s2 += __shfl_xor(s2, m, 16);
            }
            float mean = s * (1.0f / 128.0f);
            float var  = s2 * (1.0f / 128.0f) - mean * mean;
            float inv  = rsqrtf(var + 1e-5f);
            int g = row0 + r;
            if (g < rows) {
                #pragma unroll
                for (int j = 0; j < 8; ++j) {
                    int c = sub * 8 + j;
                    float y = (o_lds[r * D + c] - mean) * inv * ln_s[c] + ln_o[c];
                    if (out_f32) out_f32[(size_t)g * D + c] = y;
                    out_bf[(size_t)g * D + c] = f32_to_bf16(y);
                }
            }
        }
        __syncthreads();
    }
}

// ---------------------------------------------------------------------------
// Issue one tile's gather as 3 async B128 ops per wave (part j = sender /
// receiver / edge features; wave v covers rows 2v, 2v+1; lane -> 16B chunk).
// ---------------------------------------------------------------------------
__device__ __forceinline__ void gather_issue(int tile, unsigned ldsbase,
        const int* __restrict__ senders, const int* __restrict__ receivers,
        const unsigned short* __restrict__ n_bf,
        const unsigned short* __restrict__ e_bf, int E, int tid)
{
    const int lane = tid & 31, wv = tid >> 5;
    const int row = 2 * wv + (lane >> 4);
    const unsigned chunk = (unsigned)(lane & 15) * 16u;
    int e = tile * 16 + row;
    int ec = (e < E) ? e : 0;
    const unsigned ldsrow = ldsbase + (unsigned)row * (K1 * 2);
    {   // part 0: n[sender]
        unsigned voff = (unsigned)senders[ec] * (D * 2) + chunk;
        async_gather_b128(ldsrow + chunk, voff, (unsigned long long)n_bf);
    }
    {   // part 1: n[receiver]
        unsigned voff = (unsigned)receivers[ec] * (D * 2) + chunk;
        async_gather_b128(ldsrow + 256 + chunk, voff, (unsigned long long)n_bf);
    }
    {   // part 2: edge features
        unsigned voff = (unsigned)ec * (D * 2) + chunk;
        async_gather_b128(ldsrow + 512 + chunk, voff, (unsigned long long)e_bf);
    }
}

// ---------------------------------------------------------------------------
// Edge message MLP + scatter-add aggregation. Weights TDM-staged in LDS;
// x-gather double-buffered via async global->LDS loads.
// ---------------------------------------------------------------------------
__global__ __launch_bounds__(BLOCK)
void msg_kernel(const unsigned short* __restrict__ n_bf,
                const unsigned short* __restrict__ e_bf,
                const int* __restrict__ senders, const int* __restrict__ receivers,
                const unsigned* __restrict__ w1pk, const float* __restrict__ b1,
                const unsigned* __restrict__ w2pk, const float* __restrict__ b2,
                float* __restrict__ agg, int E)
{
    __shared__ unsigned       w1p[(K1 / 2) * H];   // 192 KB
    __shared__ unsigned       w2p[(H / 2) * D];    //  64 KB
    __shared__ unsigned short xbuf[2][16 * K1];    //  24 KB (double buffer)
    __shared__ unsigned short m1_lds[16 * H];      //   8 KB
    __shared__ float          b1_lds[H];
    __shared__ float          b2_lds[D];
    __shared__ int            rcv_lds[16];

    const int tid = threadIdx.x;
#if USE_TDM
    if (tid < 32) {
        tdm_load_1d(lds_addr_of(w1p), w1pk, (K1 / 2) * H / 2);
        tdm_load_1d(lds_addr_of(w2p), w2pk, (H / 2) * D / 2);
        __builtin_amdgcn_s_wait_tensorcnt(0);
    }
#else
    for (int p = tid; p < (K1 / 2) * H; p += BLOCK) w1p[p] = w1pk[p];
    for (int p = tid; p < (H / 2) * D; p += BLOCK)  w2p[p] = w2pk[p];
#endif
    b1_lds[tid] = b1[tid];
    if (tid < D) b2_lds[tid] = b2[tid];

    const int wave = tid >> 5, lane = tid & 31, hi = lane >> 4, mrow = lane & 15;
    const int ntiles = (E + 15) >> 4;
    const int gstride = gridDim.x;

    // prologue: prefetch first tile into buffer 0
    gather_issue(blockIdx.x, lds_addr_of(&xbuf[0][0]), senders, receivers,
                 n_bf, e_bf, E, tid);

    int parity = 0;
    for (int tile = blockIdx.x; tile < ntiles; tile += gstride, parity ^= 1) {
        if (tid < 16) {
            int e = tile * 16 + tid;
            rcv_lds[tid] = (e < E) ? receivers[e] : -1;
        }
        int nt = tile + gstride;
        if (nt < ntiles) {
            gather_issue(nt, lds_addr_of(&xbuf[parity ^ 1][0]), senders, receivers,
                         n_bf, e_bf, E, tid);
            // 3 new ops outstanding; in-order completion => current tile done
            asm volatile("s_wait_asynccnt 0x3" ::: "memory");
        } else {
            asm volatile("s_wait_asynccnt 0x0" ::: "memory");
        }
        __syncthreads();

        const unsigned short* xl = xbuf[parity];

        // GEMM1: [16,384]x[384,256] -> relu -> m1 (bf16)
        #pragma unroll
        for (int t = 0; t < 2; ++t) {
            const int col = (wave + t * 8) * 16 + (lane & 15);
            v8f acc;
            float bb = b1_lds[col];
            #pragma unroll
            for (int r = 0; r < 8; ++r) acc[r] = bb;
            #pragma unroll
            for (int kb = 0; kb < K1 / 32; ++kb) {
                const int k0 = kb * 32;
                Frag a, b;
                #pragma unroll
                for (int q = 0; q < 8; ++q) {
                    int ka = k0 + 2 * q + ((q >= 4) ? 8 : 0) + 8 * hi;
                    a.u[q] = *reinterpret_cast<const unsigned*>(&xl[mrow * K1 + ka]);
                    b.u[q] = w1p[((k0 >> 1) + q + ((q >= 4) ? 4 : 0) + 4 * hi) * H + col];
                }
                acc = __builtin_amdgcn_wmma_f32_16x16x32_bf16(false, a.v, false, b.v,
                                                              (short)0, acc, false, false);
            }
            #pragma unroll
            for (int r = 0; r < 8; ++r) {
                int row = hi ? r + 8 : r;
                m1_lds[row * H + col] = f32_to_bf16(fmaxf(acc[r], 0.0f));
            }
        }
        __syncthreads();

        // GEMM2: [16,256]x[256,128] -> relu -> atomic scatter-add
        {
            const int col = wave * 16 + (lane & 15);
            v8f acc;
            float bb = b2_lds[col];
            #pragma unroll
            for (int r = 0; r < 8; ++r) acc[r] = bb;
            #pragma unroll
            for (int kb = 0; kb < H / 32; ++kb) {
                const int k0 = kb * 32;
                Frag a, b;
                #pragma unroll
                for (int q = 0; q < 8; ++q) {
                    int ka = k0 + 2 * q + ((q >= 4) ? 8 : 0) + 8 * hi;
                    a.u[q] = *reinterpret_cast<const unsigned*>(&m1_lds[mrow * H + ka]);
                    b.u[q] = w2p[((k0 >> 1) + q + ((q >= 4) ? 4 : 0) + 4 * hi) * D + col];
                }
                acc = __builtin_amdgcn_wmma_f32_16x16x32_bf16(false, a.v, false, b.v,
                                                              (short)0, acc, false, false);
            }
            #pragma unroll
            for (int r = 0; r < 8; ++r) {
                int row = hi ? r + 8 : r;
                int rc = rcv_lds[row];
                if (rc >= 0)
                    unsafeAtomicAdd(&agg[(size_t)rc * D + col], fmaxf(acc[r], 0.0f));
            }
        }
        __syncthreads();
    }
}

// ---------------------------------------------------------------------------
// Node update: upd = LN([n, agg] @ W + b); n += upd; refresh bf16 mirror.
// ---------------------------------------------------------------------------
__global__ __launch_bounds__(BLOCK)
void upd_kernel(float* __restrict__ n, unsigned short* __restrict__ n_bf,
                const float* __restrict__ agg, const unsigned* __restrict__ wpk,
                const float* __restrict__ bias, const float* __restrict__ ln_s,
                const float* __restrict__ ln_o, int N)
{
    __shared__ unsigned       wp[(H / 2) * D];     // 64 KB
    __shared__ unsigned short a_lds[16 * H];
    __shared__ float          o_lds[16 * D];
    __shared__ float          b_lds[D];

    const int tid = threadIdx.x;
#if USE_TDM
    if (tid < 32) {
        tdm_load_1d(lds_addr_of(wp), wpk, (H / 2) * D / 2);
        __builtin_amdgcn_s_wait_tensorcnt(0);
    }
#else
    for (int p = tid; p < (H / 2) * D; p += BLOCK) wp[p] = wpk[p];
#endif
    if (tid < D) b_lds[tid] = bias[tid];
    __syncthreads();

    const int wave = tid >> 5, lane = tid & 31, hi = lane >> 4, mrow = lane & 15;
    const int col = wave * 16 + (lane & 15);
    const int ntiles = (N + 15) >> 4;

    for (int tile = blockIdx.x; tile < ntiles; tile += gridDim.x) {
        const int row0 = tile << 4;
        for (int p = tid; p < 16 * H; p += BLOCK) {
            int r = p >> 8, c = p & (H - 1);
            int g = row0 + r;
            float v = 0.0f;
            if (g < N) v = (c < D) ? n[(size_t)g * D + c] : agg[(size_t)g * D + (c - D)];
            a_lds[p] = f32_to_bf16(v);
        }
        __syncthreads();

        v8f acc;
        float bb = b_lds[col];
        #pragma unroll
        for (int r = 0; r < 8; ++r) acc[r] = bb;
        #pragma unroll
        for (int kb = 0; kb < H / 32; ++kb) {
            const int k0 = kb * 32;
            Frag a, b;
            #pragma unroll
            for (int q = 0; q < 8; ++q) {
                int ka = k0 + 2 * q + ((q >= 4) ? 8 : 0) + 8 * hi;
                a.u[q] = *reinterpret_cast<const unsigned*>(&a_lds[mrow * H + ka]);
                b.u[q] = wp[((k0 >> 1) + q + ((q >= 4) ? 4 : 0) + 4 * hi) * D + col];
            }
            acc = __builtin_amdgcn_wmma_f32_16x16x32_bf16(false, a.v, false, b.v,
                                                          (short)0, acc, false, false);
        }
        #pragma unroll
        for (int r = 0; r < 8; ++r) {
            int row = hi ? r + 8 : r;
            o_lds[row * D + col] = acc[r];
        }
        __syncthreads();

        {   // LayerNorm + residual
            int r = tid >> 4, sub = tid & 15;
            float s = 0.0f, s2 = 0.0f;
            #pragma unroll
            for (int j = 0; j < 8; ++j) {
                float v = o_lds[r * D + sub * 8 + j];
                s += v; s2 += v * v;
            }
            #pragma unroll
            for (int m = 8; m >= 1; m >>= 1) {
                s  += __shfl_xor(s,  m, 16);
                s2 += __shfl_xor(s2, m, 16);
            }
            float mean = s * (1.0f / 128.0f);
            float var  = s2 * (1.0f / 128.0f) - mean * mean;
            float inv  = rsqrtf(var + 1e-5f);
            int g = row0 + r;
            if (g < N) {
                #pragma unroll
                for (int j = 0; j < 8; ++j) {
                    int c = sub * 8 + j;
                    float u = (o_lds[r * D + c] - mean) * inv * ln_s[c] + ln_o[c];
                    float y = n[(size_t)g * D + c] + u;
                    n[(size_t)g * D + c] = y;
                    n_bf[(size_t)g * D + c] = f32_to_bf16(y);
                }
            }
        }
        __syncthreads();
    }
}

__global__ void zero_kernel(float* __restrict__ p, size_t cnt) {
    size_t i = (size_t)blockIdx.x * blockDim.x + threadIdx.x;
    size_t stride = (size_t)gridDim.x * blockDim.x;
    for (; i < cnt; i += stride) p[i] = 0.0f;
}

// ---------------------------------------------------------------------------
extern "C" void kernel_launch(void* const* d_in, const int* in_sizes, int n_in,
                              void* d_out, int out_size, void* d_ws, size_t ws_size,
                              hipStream_t stream)
{
    (void)n_in; (void)out_size; (void)ws_size;
    const float* nodes    = (const float*)d_in[0];
    const float* edges    = (const float*)d_in[1];
    const int*   senders  = (const int*)d_in[2];
    const int*   receivers= (const int*)d_in[3];
    const float* w_node   = (const float*)d_in[4];
    const float* b_node   = (const float*)d_in[5];
    const float* w_edge   = (const float*)d_in[6];
    const float* b_edge   = (const float*)d_in[7];
    const float* ln_n_s   = (const float*)d_in[8];
    const float* ln_n_o   = (const float*)d_in[9];
    const float* ln_e_s   = (const float*)d_in[10];
    const float* ln_e_o   = (const float*)d_in[11];
    const float* msg_w1   = (const float*)d_in[12];
    const float* msg_b1   = (const float*)d_in[13];
    const float* msg_w2   = (const float*)d_in[14];
    const float* msg_b2   = (const float*)d_in[15];
    const float* upd_w    = (const float*)d_in[16];
    const float* upd_b    = (const float*)d_in[17];
    const float* ln_s     = (const float*)d_in[18];
    const float* ln_o     = (const float*)d_in[19];

    const int N = in_sizes[0] / D;
    const int E = in_sizes[1] / D;

    char* ws = (char*)d_ws;
    auto align_up = [](size_t x) { return (x + 255) & ~(size_t)255; };
    size_t off = 0;
    unsigned short* n_bf = (unsigned short*)(ws + off);
    off += align_up((size_t)N * D * sizeof(unsigned short));
    float* agg = (float*)(ws + off);
    off += align_up((size_t)N * D * sizeof(float));
    unsigned short* e_bf = (unsigned short*)(ws + off);
    off += align_up((size_t)E * D * sizeof(unsigned short));
    unsigned* wnode_p = (unsigned*)(ws + off); off += align_up((size_t)(D/2)*D*4);
    unsigned* wedge_p = (unsigned*)(ws + off); off += align_up((size_t)(D/2)*D*4);
    unsigned* w1p_g[3]; unsigned* w2p_g[3]; unsigned* wup_g[3];
    for (int l = 0; l < 3; ++l) {
        w1p_g[l] = (unsigned*)(ws + off); off += align_up((size_t)(K1/2)*H*4);
        w2p_g[l] = (unsigned*)(ws + off); off += align_up((size_t)(H/2)*D*4);
        wup_g[l] = (unsigned*)(ws + off); off += align_up((size_t)(H/2)*D*4);
    }

    float* n = (float*)d_out;   // node features live in d_out, updated in place

    // ---- pack all weights once (bf16 pairs) ----
    pack_weights_kernel<<<32, 256, 0, stream>>>(w_node, wnode_p, (D/2)*D, D);
    pack_weights_kernel<<<32, 256, 0, stream>>>(w_edge, wedge_p, (D/2)*D, D);
    for (int l = 0; l < 3; ++l) {
        pack_weights_kernel<<<192, 256, 0, stream>>>(msg_w1 + (size_t)l*K1*H, w1p_g[l], (K1/2)*H, H);
        pack_weights_kernel<<<64, 256, 0, stream>>>(msg_w2 + (size_t)l*H*D,  w2p_g[l], (H/2)*D, D);
        pack_weights_kernel<<<64, 256, 0, stream>>>(upd_w  + (size_t)l*H*D,  wup_g[l], (H/2)*D, D);
    }

    const int ntile_n = (N + 15) / 16;
    const int ntile_e = (E + 15) / 16;
    const int g_n   = ntile_n < 2048 ? ntile_n : 2048;
    const int g_e   = ntile_e < 4096 ? ntile_e : 4096;
    const int g_msg = ntile_e < 2048 ? ntile_e : 2048;

    embed_ln_kernel<<<g_n, BLOCK, 0, stream>>>(nodes, wnode_p, b_node,
                                               ln_n_s, ln_n_o, n, n_bf, N);
    embed_ln_kernel<<<g_e, BLOCK, 0, stream>>>(edges, wedge_p, b_edge,
                                               ln_e_s, ln_e_o, nullptr, e_bf, E);
    for (int l = 0; l < 3; ++l) {
        zero_kernel<<<1024, 256, 0, stream>>>(agg, (size_t)N * D);
        msg_kernel<<<g_msg, BLOCK, 0, stream>>>(n_bf, e_bf, senders, receivers,
            w1p_g[l], msg_b1 + (size_t)l * H,
            w2p_g[l], msg_b2 + (size_t)l * D,
            agg, E);
        upd_kernel<<<g_n, BLOCK, 0, stream>>>(n, n_bf, agg,
            wup_g[l], upd_b + (size_t)l * D,
            ln_s + (size_t)l * D, ln_o + (size_t)l * D, N);
    }
}
